// GraphTemporalRefiner_13778255086181
// MI455X (gfx1250) — compile-verified, matrix operand order checked by reading, not run on hardware
//
#include <hip/hip_runtime.h>
#include <hip/hip_bf16.h>
#include <math.h>

// ---------------------------------------------------------------------------
// GraphTemporalRefiner for MI455X (gfx1250, wave32, WMMA + async-to-LDS).
//
// Reference output = MLP(LN(out_proj(o[:, -1, :]))); with unmasked softmax,
// o[:, -1] needs only q at t=T-1 but K,V at all t.  Pipeline:
//   1) h = x @ theta_w^T + theta_b            (WMMA bf16, M=32768 N=128 K=64)
//   2) g = row-normalized banded adj conv(h)  (VALU, stored bf16)
//   3) K,V = g @ Wk^T, g @ Wv^T               (WMMA bf16, M=32768 N=128 K=128 x2)
//   4) per (b, head): q = Wq g_last ; softmax(q.K/sqrt(32)) ; o = attn.V
//   5) out_proj(last) -> LayerNorm -> GELU MLP -> (B, 2)
//
// The shared A tile of each block (16 rows of x / g) is staged into LDS once
// via GLOBAL_LOAD_ASYNC_TO_LDS_B128 (ASYNCcnt) instead of being fetched 8x
// (once per wave) from global.
// ---------------------------------------------------------------------------

#define B_   32
#define T_   1024
#define DIN  64
#define H_   128
#define NH_  4
#define HD_  32
#define RAD  3

typedef __bf16 bf16_t;
typedef __attribute__((ext_vector_type(16))) __bf16 v16bf;
typedef __attribute__((ext_vector_type(8)))  float  v8f;

// LDS byte offset of a __shared__ object (generic -> addrspace(3) cast).
typedef __attribute__((address_space(3))) const void lds_cvoid;
__device__ __forceinline__ unsigned lds_off_of(const void* p) {
    return (unsigned)(size_t)(lds_cvoid*)p;
}

// Issue one per-lane 16-byte async global->LDS transfer (ASYNCcnt-tracked).
__device__ __forceinline__ void async_copy_b128(unsigned lds_byte_off,
                                                const void* gaddr) {
    asm volatile("global_load_async_to_lds_b128 %0, %1, off"
                 :: "v"(lds_byte_off), "v"(gaddr)
                 : "memory");
}
__device__ __forceinline__ void wait_asynccnt0() {
    asm volatile("s_wait_asynccnt 0x0" ::: "memory");
}

// K offset inside a 16x16x32 bf16 fragment: element e, lane-group grp (0/1).
// Lanes 0-15: e0..7 -> K 0..7, e8..15 -> K 16..23.  Lanes 16-31: +8.
__device__ __forceinline__ int wmma_k(int e, int grp) {
    return (e & 7) + ((e >> 3) << 4) + (grp << 3);
}

// ---------------------------------------------------------------------------
// Kernel 1: h[m, n] = sum_k x[m, k] * theta_w[n, k] + theta_b[n]
//   M = B*T = 32768, N = H = 128, K = DIN = 64.  One 16x16 tile per wave,
//   8 waves/block cover N, grid.x covers M/16.  A tile staged via async-LDS.
// ---------------------------------------------------------------------------
__global__ void __launch_bounds__(256) h_gemm_kernel(
    const float* __restrict__ x, const float* __restrict__ tw,
    const float* __restrict__ tb, float* __restrict__ h)
{
    __shared__ float xtile[16 * DIN];            // 16 rows x 64 f32 = 4 KB

    const int tid  = threadIdx.x;
    const int lane = tid & 31;
    const int wave = tid >> 5;
    const int grp  = lane >> 4;
    const int l16  = lane & 15;
    const int m0   = blockIdx.x * 16;
    const int n0   = wave * 16;
    const int bcol = n0 + l16;                   // B column this lane loads

    // Stage A tile: 16 rows x 256 B = 256 lanes x 16 B (one b128 per lane).
    {
        const int row = tid >> 4, chunk = tid & 15;
        const float* src = x + (size_t)(m0 + row) * DIN + chunk * 4;
        async_copy_b128(lds_off_of(&xtile[row * DIN + chunk * 4]), src);
    }
    wait_asynccnt0();
    __syncthreads();

    v8f c;
    const float bias = tb[bcol];
#pragma unroll
    for (int r = 0; r < 8; ++r) c[r] = bias;

#pragma unroll
    for (int kk = 0; kk < DIN; kk += 32) {
        v16bf a, bm;
#pragma unroll
        for (int e = 0; e < 16; ++e) {
            const int k = kk + wmma_k(e, grp);
            a[e]  = (bf16_t)xtile[l16 * DIN + k];
            bm[e] = (bf16_t)tw[(size_t)bcol * DIN + k];
        }
        c = __builtin_amdgcn_wmma_f32_16x16x32_bf16(
                false, a, false, bm, (short)0, c, false, false);
    }
#pragma unroll
    for (int r = 0; r < 8; ++r)
        h[(size_t)(m0 + 8 * grp + r) * H_ + n0 + l16] = c[r];
}

// ---------------------------------------------------------------------------
// Kernel 2: banded, row-normalized adjacency:
//   g[b,i,:] = sum_{|d|<=3} (1/(1+|d|))/norm_i * h[b,i+d,:]   (stored bf16)
// ---------------------------------------------------------------------------
__global__ void __launch_bounds__(256) band_conv_kernel(
    const float* __restrict__ h, bf16_t* __restrict__ g)
{
    const int idx = blockIdx.x * 256 + threadIdx.x;   // [0, B*T*H)
    const int hh = idx & (H_ - 1);
    const int bt = idx >> 7;                          // H_ == 128
    const int t  = bt & (T_ - 1);

    float acc = 0.f, norm = 0.f;
#pragma unroll
    for (int d = -RAD; d <= RAD; ++d) {
        const int j = t + d;
        if (j >= 0 && j < T_) {
            const float w = 1.0f / (1.0f + (float)(d < 0 ? -d : d));
            norm += w;
            acc  += w * h[(size_t)(bt + d) * H_ + hh];
        }
    }
    g[idx] = (bf16_t)(acc / fmaxf(norm, 1e-8f));
}

// ---------------------------------------------------------------------------
// Kernel 3: K = g @ Wk^T + bk ; V = g @ Wv^T + bv  (rows H..2H / 2H..3H of
//   in_proj).  M = 32768, N = 128 each, K = 128 (4 WMMA steps).  A tile
//   staged once via async-LDS (bf16, no conversion) and reused for both
//   halves by all 8 waves.
// ---------------------------------------------------------------------------
__global__ void __launch_bounds__(256) kv_gemm_kernel(
    const bf16_t* __restrict__ g, const float* __restrict__ ipw,
    const float* __restrict__ ipb, float* __restrict__ kbuf,
    float* __restrict__ vbuf)
{
    __shared__ bf16_t gtile[16 * H_];            // 16 rows x 128 bf16 = 4 KB

    const int tid  = threadIdx.x;
    const int lane = tid & 31;
    const int wave = tid >> 5;
    const int grp  = lane >> 4;
    const int l16  = lane & 15;
    const int m0   = blockIdx.x * 16;
    const int n0   = wave * 16;
    const int bcol = n0 + l16;

    // Stage A tile: 16 rows x 256 B = 256 lanes x 16 B (one b128 per lane).
    {
        const int row = tid >> 4, chunk = tid & 15;
        const bf16_t* src = g + (size_t)(m0 + row) * H_ + chunk * 8;
        async_copy_b128(lds_off_of(&gtile[row * H_ + chunk * 8]), src);
    }
    wait_asynccnt0();
    __syncthreads();

    v16bf afr[4];
#pragma unroll
    for (int kb = 0; kb < 4; ++kb)
#pragma unroll
        for (int e = 0; e < 16; ++e)
            afr[kb][e] = gtile[l16 * H_ + kb * 32 + wmma_k(e, grp)];

#pragma unroll
    for (int half = 0; half < 2; ++half) {
        const int wr = H_ * (1 + half) + bcol;       // row of in_proj_w
        const float* wrow = ipw + (size_t)wr * H_;
        v8f c;
        const float bias = ipb[wr];
#pragma unroll
        for (int r = 0; r < 8; ++r) c[r] = bias;

#pragma unroll
        for (int kb = 0; kb < 4; ++kb) {
            v16bf bm;
#pragma unroll
            for (int e = 0; e < 16; ++e)
                bm[e] = (bf16_t)wrow[kb * 32 + wmma_k(e, grp)];
            c = __builtin_amdgcn_wmma_f32_16x16x32_bf16(
                    false, afr[kb], false, bm, (short)0, c, false, false);
        }
        float* dst = half ? vbuf : kbuf;
#pragma unroll
        for (int r = 0; r < 8; ++r)
            dst[(size_t)(m0 + 8 * grp + r) * H_ + n0 + l16] = c[r];
    }
}

// ---------------------------------------------------------------------------
// Kernel 4: last-token attention, one block per (batch, head).
//   q = Wq . g[b, T-1, :], scores = q.K/sqrt(HD), softmax, o = attn.V
// ---------------------------------------------------------------------------
__global__ void __launch_bounds__(256) attn_kernel(
    const bf16_t* __restrict__ g, const float* __restrict__ ipw,
    const float* __restrict__ ipb, const float* __restrict__ kbuf,
    const float* __restrict__ vbuf, float* __restrict__ obuf)
{
    const int b   = blockIdx.x / NH_;
    const int n   = blockIdx.x % NH_;
    const int tid = threadIdx.x;

    __shared__ float glast[H_];
    __shared__ float qs[HD_];
    __shared__ float sc[T_];
    __shared__ float rbuf[256];
    __shared__ float part[8][HD_];
    __shared__ float smax, ssum;

    for (int i = tid; i < H_; i += 256)
        glast[i] = (float)g[(size_t)(b * T_ + (T_ - 1)) * H_ + i];
    __syncthreads();

    if (tid < HD_) {
        const int r = n * HD_ + tid;                 // row in Wq block
        const float* wrow = ipw + (size_t)r * H_;
        float acc = ipb[r];
        for (int k2 = 0; k2 < H_; ++k2) acc += wrow[k2] * glast[k2];
        qs[tid] = acc * 0.17677669529663689f;        // 1/sqrt(32)
    }
    __syncthreads();

    float lmax = -3.402823466e38f;
    for (int s = tid; s < T_; s += 256) {
        const float* kr = kbuf + (size_t)(b * T_ + s) * H_ + n * HD_;
        float acc = 0.f;
#pragma unroll
        for (int d = 0; d < HD_; ++d) acc += qs[d] * kr[d];
        sc[s] = acc;
        lmax = fmaxf(lmax, acc);
    }
    rbuf[tid] = lmax;
    __syncthreads();
    for (int off = 128; off > 0; off >>= 1) {
        if (tid < off) rbuf[tid] = fmaxf(rbuf[tid], rbuf[tid + off]);
        __syncthreads();
    }
    if (tid == 0) smax = rbuf[0];
    __syncthreads();

    float lsum = 0.f;
    for (int s = tid; s < T_; s += 256) {
        const float e = __expf(sc[s] - smax);
        sc[s] = e;
        lsum += e;
    }
    rbuf[tid] = lsum;
    __syncthreads();
    for (int off = 128; off > 0; off >>= 1) {
        if (tid < off) rbuf[tid] += rbuf[tid + off];
        __syncthreads();
    }
    if (tid == 0) ssum = rbuf[0];
    __syncthreads();

    const int d  = tid & (HD_ - 1);
    const int sg = tid >> 5;                         // 8 s-groups
    float acc = 0.f;
    for (int s = sg; s < T_; s += 8)
        acc += sc[s] * vbuf[(size_t)(b * T_ + s) * H_ + n * HD_ + d];
    part[sg][d] = acc;
    __syncthreads();
    if (tid < HD_) {
        float o = 0.f;
#pragma unroll
        for (int j = 0; j < 8; ++j) o += part[j][tid];
        obuf[b * H_ + n * HD_ + tid] = o / ssum;
    }
}

// ---------------------------------------------------------------------------
// Kernel 5: out_proj -> LayerNorm -> GELU(exact) MLP -> (B, 2)
// ---------------------------------------------------------------------------
__global__ void __launch_bounds__(128) head_kernel(
    const float* __restrict__ obuf, const float* __restrict__ opw,
    const float* __restrict__ opb,  const float* __restrict__ lng,
    const float* __restrict__ lnb,  const float* __restrict__ w1,
    const float* __restrict__ b1,   const float* __restrict__ w2,
    const float* __restrict__ b2,   float* __restrict__ out)
{
    const int b = blockIdx.x;
    const int j = threadIdx.x;
    __shared__ float a_s[H_], ln_s[H_], hid[H_];
    __shared__ float mu, rstd;

    const float* orow = obuf + b * H_;
    const float* wrow = opw + (size_t)j * H_;
    float acc = opb[j];
    for (int i = 0; i < H_; ++i) acc += wrow[i] * orow[i];
    a_s[j] = acc;
    __syncthreads();

    if (j == 0) {
        float s = 0.f;
        for (int i = 0; i < H_; ++i) s += a_s[i];
        const float m = s / H_;
        float s2 = 0.f;
        for (int i = 0; i < H_; ++i) { const float dd = a_s[i] - m; s2 += dd * dd; }
        mu = m;
        rstd = rsqrtf(s2 / H_ + 1e-5f);
    }
    __syncthreads();

    ln_s[j] = (a_s[j] - mu) * rstd * lng[j] + lnb[j];
    __syncthreads();

    const float* w1row = w1 + (size_t)j * H_;
    float h1 = b1[j];
    for (int i = 0; i < H_; ++i) h1 += w1row[i] * ln_s[i];
    hid[j] = 0.5f * h1 * (1.0f + erff(h1 * 0.70710678118654752f));
    __syncthreads();

    if (j < 2) {
        const float* w2row = w2 + (size_t)j * H_;
        float o = b2[j];
        for (int i = 0; i < H_; ++i) o += w2row[i] * hid[i];
        out[b * 2 + j] = o;
    }
}

// ---------------------------------------------------------------------------
extern "C" void kernel_launch(void* const* d_in, const int* in_sizes, int n_in,
                              void* d_out, int out_size, void* d_ws, size_t ws_size,
                              hipStream_t stream)
{
    (void)in_sizes; (void)n_in; (void)out_size; (void)ws_size;

    const float* x   = (const float*)d_in[0];
    const float* tw  = (const float*)d_in[1];
    const float* tb  = (const float*)d_in[2];
    const float* ipw = (const float*)d_in[3];
    const float* ipb = (const float*)d_in[4];
    const float* opw = (const float*)d_in[5];
    const float* opb = (const float*)d_in[6];
    const float* lng = (const float*)d_in[7];
    const float* lnb = (const float*)d_in[8];
    const float* w1  = (const float*)d_in[9];
    const float* b1  = (const float*)d_in[10];
    const float* w2  = (const float*)d_in[11];
    const float* b2  = (const float*)d_in[12];
    float* out = (float*)d_out;

    // Workspace layout (all intermediates L2-resident: 192 MB L2 >> 56 MB):
    //   [ 0MB) h    f32  B*T*H   (16 MB)
    //   [16MB) g    bf16 B*T*H   ( 8 MB)
    //   [24MB) K    f32  B*T*H   (16 MB)
    //   [40MB) V    f32  B*T*H   (16 MB)
    //   [56MB) o    f32  B*H     (16 KB)
    char* ws = (char*)d_ws;
    float*  hbuf = (float*)(ws);
    bf16_t* gbuf = (bf16_t*)(ws + (size_t)16 * 1024 * 1024);
    float*  kbuf = (float*)(ws + (size_t)24 * 1024 * 1024);
    float*  vbuf = (float*)(ws + (size_t)40 * 1024 * 1024);
    float*  obuf = (float*)(ws + (size_t)56 * 1024 * 1024);

    hipLaunchKernelGGL(h_gemm_kernel, dim3((B_ * T_) / 16), dim3(256), 0, stream,
                       x, tw, tb, hbuf);
    hipLaunchKernelGGL(band_conv_kernel, dim3((B_ * T_ * H_) / 256), dim3(256), 0, stream,
                       hbuf, gbuf);
    hipLaunchKernelGGL(kv_gemm_kernel, dim3((B_ * T_) / 16), dim3(256), 0, stream,
                       gbuf, ipw, ipb, kbuf, vbuf);
    hipLaunchKernelGGL(attn_kernel, dim3(B_ * NH_), dim3(256), 0, stream,
                       gbuf, ipw, ipb, kbuf, vbuf, obuf);
    hipLaunchKernelGGL(head_kernel, dim3(B_), dim3(128), 0, stream,
                       obuf, opw, opb, lng, lnb, w1, b1, w2, b2, out);
}
